// ChiStream_71055938945316
// MI455X (gfx1250) — compile-verified, matrix operand order checked by reading, not run on hardware
//
#include <hip/hip_runtime.h>
#include <hip/hip_bf16.h>

// ============================================================================
// ChiStream forward for MI455X (gfx1250, wave32, WMMA).
// All heavy math (18 x [tokens x 512 x 512] linears + batched P*V apply) runs
// on v_wmma_f32_16x16x32_f16 with fp32 accumulation. Activations are fp32 in
// a merged [B,T,S,512] layout; f32->f16 conversion happens while staging to
// LDS (vectorized, branch-free). Weights are converted to f16 once per call.
// Softmax rows are zero-padded to multiples of 32 so the WMMA apply kernel
// needs no K guards. Epilogues transpose accumulators through wave-private
// LDS slabs so all global stores are coalesced b128.
// Requires ~650 MB of workspace (d_ws).
// ============================================================================

typedef _Float16 v16h __attribute__((ext_vector_type(16)));
typedef _Float16 v8h  __attribute__((ext_vector_type(8)));
typedef float    v8f  __attribute__((ext_vector_type(8)));

#define BSZ 16
#define NH  8

// ---------------------------------------------------------------------------
// Pack [B,H,T,S,64] (+optional concat along T or S) -> merged [B,T,S,512] f32
// ---------------------------------------------------------------------------
__global__ __launch_bounds__(256) void pack_merge(
    const float* __restrict__ a, const float* __restrict__ b2,
    float* __restrict__ out, int T, int S, int splitS, int splitAt,
    long long total4)
{
  long long idx = (long long)blockIdx.x * 256 + threadIdx.x;
  if (idx >= total4) return;
  int c4 = (int)(idx & 127);
  long long e = idx >> 7;
  int s = (int)(e % S); e /= S;
  int t = (int)(e % T); int b = (int)(e / T);
  int h = c4 >> 4, cc = c4 & 15;
  const float* src; int TT, SS, tt = t, ss = s;
  if (splitS) {
    TT = T;
    if (s < splitAt) { src = a;  SS = splitAt; }
    else             { src = b2; SS = S - splitAt; ss = s - splitAt; }
  } else {
    SS = S;
    if (t < splitAt) { src = a;  TT = splitAt; }
    else             { src = b2; TT = T - splitAt; tt = t - splitAt; }
  }
  const float4* sp = (const float4*)src;
  float4 v = sp[((((long long)b * NH + h) * TT + tt) * SS + ss) * 16 + cc];
  ((float4*)out)[idx] = v;
}

// ---------------------------------------------------------------------------
// Copy merged tokens into a (possibly larger) merged buffer: concat helper
// ---------------------------------------------------------------------------
__global__ __launch_bounds__(256) void copy_tokens(
    const float* __restrict__ in, float* __restrict__ out,
    int T, int S, int Td, int Sd, int t0, int s0, long long total4)
{
  long long idx = (long long)blockIdx.x * 256 + threadIdx.x;
  if (idx >= total4) return;
  int c4 = (int)(idx & 127);
  long long e = idx >> 7;
  int s = (int)(e % S); e /= S;
  int t = (int)(e % T); int b = (int)(e / T);
  ((float4*)out)[(((long long)b * Td + t0 + t) * Sd + (s0 + s)) * 128 + c4] =
      ((const float4*)in)[idx];
}

// ---------------------------------------------------------------------------
// f32 -> f16 weight conversion (512x512 each)
// ---------------------------------------------------------------------------
__global__ __launch_bounds__(256) void cvt_f16(const float* __restrict__ in,
                                               _Float16* __restrict__ out, int n)
{
  int i = blockIdx.x * 256 + threadIdx.x;
  if (i < n) out[i] = (_Float16)in[i];
}

// ---------------------------------------------------------------------------
// GEMM: C[M,512] = A[M,512] * W^T + bias.  W is [512(out),512(in)] f16.
// M must be a multiple of 128 (all call sites guarantee this).
// Block tile 128x64, 8 waves in 4x2 grid, each wave 2x2 of 16x16 WMMA tiles,
// K stepped by 32 through LDS. Epilogue goes through a wave-private LDS slab
// so global stores are coalesced float4 (4 rows x 128B per instruction).
// ---------------------------------------------------------------------------
__global__ __launch_bounds__(256) void gemm512(
    const float* __restrict__ A, const _Float16* __restrict__ W,
    const float* __restrict__ bias, float* __restrict__ C, int M)
{
  __shared__ _Float16 lA[128][40];
  __shared__ _Float16 lB[64][40];
  __shared__ float sOut[8][32][36];   // per-wave 32x32 f32 tile, 16B-mult rows
  int tid = threadIdx.x;
  int wave = tid >> 5, lane = tid & 31, r = lane & 15, hl = lane >> 4;
  int waveM = wave >> 1, waveN = wave & 1;
  int m0 = blockIdx.x * 128, n0 = blockIdx.y * 64;
  (void)M;

  v8f acc[2][2] = {};
  for (int k0 = 0; k0 < 512; k0 += 32) {
    { // stage A: 128 rows x 32 cols, 2 threads/row, f32->f16, 16B LDS stores
      int row = tid >> 1, col = (tid & 1) * 16;
      const float4* src = (const float4*)(A + (size_t)(m0 + row) * 512 + k0 + col);
      float4 f0 = src[0], f1 = src[1], f2 = src[2], f3 = src[3];
      if (k0 < 480) __builtin_prefetch((const char*)src + 128, 0, 3);
      v8h h0, h1;
      h0[0] = (_Float16)f0.x; h0[1] = (_Float16)f0.y;
      h0[2] = (_Float16)f0.z; h0[3] = (_Float16)f0.w;
      h0[4] = (_Float16)f1.x; h0[5] = (_Float16)f1.y;
      h0[6] = (_Float16)f1.z; h0[7] = (_Float16)f1.w;
      h1[0] = (_Float16)f2.x; h1[1] = (_Float16)f2.y;
      h1[2] = (_Float16)f2.z; h1[3] = (_Float16)f2.w;
      h1[4] = (_Float16)f3.x; h1[5] = (_Float16)f3.y;
      h1[6] = (_Float16)f3.z; h1[7] = (_Float16)f3.w;
      *(v8h*)&lA[row][col]     = h0;
      *(v8h*)&lA[row][col + 8] = h1;
    }
    { // stage W tile: 64 rows x 32 cols, one b128 load + b128 store per thread
      int row = tid >> 2, col = (tid & 3) * 8;
      const v8h* src = (const v8h*)(W + (size_t)(n0 + row) * 512 + k0 + col);
      *(v8h*)&lB[row][col] = *src;
    }
    __syncthreads();
#pragma unroll
    for (int tm = 0; tm < 2; ++tm) {
      v16h av;
      int ar = waveM * 32 + tm * 16 + r;
#pragma unroll
      for (int j = 0; j < 8; ++j) av[j]     = lA[ar][hl * 8 + j];
#pragma unroll
      for (int j = 0; j < 8; ++j) av[8 + j] = lA[ar][16 + hl * 8 + j];
#pragma unroll
      for (int tn = 0; tn < 2; ++tn) {
        v16h bv;
        int br = waveN * 32 + tn * 16 + r;
#pragma unroll
        for (int j = 0; j < 8; ++j) bv[j]     = lB[br][hl * 8 + j];
#pragma unroll
        for (int j = 0; j < 8; ++j) bv[8 + j] = lB[br][16 + hl * 8 + j];
        acc[tm][tn] = __builtin_amdgcn_wmma_f32_16x16x32_f16(
            false, av, false, bv, (short)0, acc[tm][tn], false, false);
      }
    }
    __syncthreads();
  }
  // ---- epilogue: transpose through wave-private LDS slab ----
  // D layout: lanes 0-15 -> n=r, VGPR g -> m=g; lanes 16-31 -> m=8+g.
  // LDS ops from one wave execute in order (ISA 7.3), and each wave touches
  // only its own slab, so no barrier is required between store and load.
#pragma unroll
  for (int tm = 0; tm < 2; ++tm)
#pragma unroll
    for (int tn = 0; tn < 2; ++tn)
#pragma unroll
      for (int g = 0; g < 8; ++g)
        sOut[wave][tm * 16 + hl * 8 + g][tn * 16 + r] = acc[tm][tn][g];
  {
    int wm = m0 + waveM * 32, wn = n0 + waveN * 32;
    int c4 = lane & 7, r8 = lane >> 3;          // 8 lanes/row, 4 rows/iter
    float4 bb = ((const float4*)(bias + wn))[c4];
#pragma unroll
    for (int it = 0; it < 8; ++it) {
      int row = it * 4 + r8;
      float4 v = *(const float4*)&sOut[wave][row][c4 * 4];
      v.x += bb.x; v.y += bb.y; v.z += bb.z; v.w += bb.w;
      *(float4*)(C + (size_t)(wm + row) * 512 + wn + c4 * 4) = v;
    }
  }
}

// ---------------------------------------------------------------------------
// Mean-pool merged [B,T,S,512] over T (overT=1 -> out [B,S,512]) or over S
// ---------------------------------------------------------------------------
__global__ __launch_bounds__(256) void pool_mean(
    const float* __restrict__ in, float* __restrict__ out,
    int T, int S, int overT, long long total4)
{
  long long idx = (long long)blockIdx.x * 256 + threadIdx.x;
  if (idx >= total4) return;
  int c4 = (int)(idx & 127);
  long long e = idx >> 7;
  int L = overT ? S : T;
  int l = (int)(e % L);
  int b = (int)(e / L);
  const float4* ip = (const float4*)in;
  float sx = 0.f, sy = 0.f, sz = 0.f, sw = 0.f;
  if (overT) {
    for (int t = 0; t < T; ++t) {
      float4 v = ip[(((long long)b * T + t) * S + l) * 128 + c4];
      sx += v.x; sy += v.y; sz += v.z; sw += v.w;
    }
  } else {
    for (int s = 0; s < S; ++s) {
      float4 v = ip[(((long long)b * T + l) * S + s) * 128 + c4];
      sx += v.x; sy += v.y; sz += v.z; sw += v.w;
    }
  }
  float sc = 1.f / (overT ? T : S);
  float4 o; o.x = sx * sc; o.y = sy * sc; o.z = sz * sc; o.w = sw * sc;
  ((float4*)out)[((long long)b * L + l) * 128 + c4] = o;
}

// ---------------------------------------------------------------------------
// Scores + softmax: one block per (b,h,i). qp [B,Lq,512], kp [B,Lk,512].
// Row stride is Lk32 (Lk padded to x32); pad columns written as 0 so the
// WMMA apply kernel can read unguarded 32-wide K tiles.
// ---------------------------------------------------------------------------
__global__ __launch_bounds__(256) void attn_scores(
    const float* __restrict__ qp, const float* __restrict__ kp,
    float* __restrict__ P, int Lq, int Lk, int Lk32)
{
  int bx = blockIdx.x;
  int i = bx % Lq; bx /= Lq;
  int h = bx % NH; int b = bx / NH;
  __shared__ float qrow[64];
  __shared__ float sm[256];
  int t = threadIdx.x;
  if (t < 64) qrow[t] = qp[((size_t)b * Lq + i) * 512 + h * 64 + t];
  __syncthreads();
  float sc = -3.4e38f;
  if (t < Lk) {
    const float* kr = kp + ((size_t)b * Lk + t) * 512 + h * 64;
    float d = 0.f;
#pragma unroll 8
    for (int c = 0; c < 64; ++c) d += qrow[c] * kr[c];
    sc = d * 0.125f;
  }
  sm[t] = sc; __syncthreads();
  for (int o = 128; o > 0; o >>= 1) {
    if (t < o) sm[t] = fmaxf(sm[t], sm[t + o]);
    __syncthreads();
  }
  float mx = sm[0]; __syncthreads();
  float ex = (t < Lk) ? __expf(sc - mx) : 0.f;
  sm[t] = ex; __syncthreads();
  for (int o = 128; o > 0; o >>= 1) {
    if (t < o) sm[t] += sm[t + o];
    __syncthreads();
  }
  float inv = 1.f / sm[0];
  if (t < Lk32)
    P[((size_t)(b * NH + h) * Lq + i) * Lk32 + t] = (t < Lk) ? ex * inv : 0.f;
}

// ---------------------------------------------------------------------------
// Apply: O[m, c] = sum_lk P[b,h,m,lk] * V[lk, c]  (c = 64 head channels),
// batched over (b, group g, head h) with arbitrary element strides so the
// same kernel serves both streams. WMMA 16x16x32. V tile is staged to LDS
// TRANSPOSED ([c][lk]) so B operands are contiguous ds_load_b128 pairs.
// P rows are zero-padded to Lk32, so A operands are unguarded float4 loads.
// Epilogue transposes each 16x64 tile through a wave-private LDS slab for
// coalesced float4 stores.
// ---------------------------------------------------------------------------
__global__ __launch_bounds__(256) void attn_apply(
    const float* __restrict__ P, const float* __restrict__ V,
    float* __restrict__ O, int G, int Lq, int Lk, int Lk32,
    long long vOffB, long long vOffG, long long vK,
    long long oOffB, long long oOffG, long long oM)
{
  int bx = blockIdx.x;
  int h = bx % NH; bx /= NH;
  int g = bx % G;  int b = bx / G;
  const float* Pb = P + (size_t)(b * NH + h) * Lq * Lk32;
  const float* Vb = V + (size_t)b * vOffB + (size_t)g * vOffG + h * 64;
  float* Ob = O + (size_t)b * oOffB + (size_t)g * oOffG + h * 64;

  __shared__ _Float16 lVt[64][40];     // transposed V tile: [channel][lk]
  __shared__ float sOut[8][16][68];    // per-wave 16x64 f32 tile
  int tid = threadIdx.x, wave = tid >> 5, lane = tid & 31;
  int r = lane & 15, hl = lane >> 4;
  int ntiles = (Lq + 15) >> 4;         // <= 16

  v8f acc[2][4] = {};
  for (int k0 = 0; k0 < Lk; k0 += 32) {
    { // stage V^T: thread -> channel c, 8 consecutive lk; 16B LDS store
      int c = tid >> 2, lkb = (tid & 3) * 8;
      const float* src = Vb + c;
      v8h hv;
      if (k0 + 32 <= Lk) {
#pragma unroll
        for (int j = 0; j < 8; ++j)
          hv[j] = (_Float16)src[(size_t)(k0 + lkb + j) * vK];
      } else {
#pragma unroll
        for (int j = 0; j < 8; ++j)
          hv[j] = (k0 + lkb + j < Lk)
                      ? (_Float16)src[(size_t)(k0 + lkb + j) * vK]
                      : (_Float16)0.f;
      }
      *(v8h*)&lVt[c][lkb] = hv;
    }
    __syncthreads();
#pragma unroll
    for (int ai = 0; ai < 2; ++ai) {
      int mt = wave + ai * 8;
      if (mt < ntiles) {
        int lq = mt * 16 + r;
        int lqc = (lq < Lq) ? lq : (Lq - 1);   // rows >= Lq never stored
        const float* prow = Pb + (size_t)lqc * Lk32 + k0;
        float4 f0 = ((const float4*)(prow + hl * 8))[0];
        float4 f1 = ((const float4*)(prow + hl * 8))[1];
        float4 f2 = ((const float4*)(prow + 16 + hl * 8))[0];
        float4 f3 = ((const float4*)(prow + 16 + hl * 8))[1];
        v16h av;
        av[0]  = (_Float16)f0.x; av[1]  = (_Float16)f0.y;
        av[2]  = (_Float16)f0.z; av[3]  = (_Float16)f0.w;
        av[4]  = (_Float16)f1.x; av[5]  = (_Float16)f1.y;
        av[6]  = (_Float16)f1.z; av[7]  = (_Float16)f1.w;
        av[8]  = (_Float16)f2.x; av[9]  = (_Float16)f2.y;
        av[10] = (_Float16)f2.z; av[11] = (_Float16)f2.w;
        av[12] = (_Float16)f3.x; av[13] = (_Float16)f3.y;
        av[14] = (_Float16)f3.z; av[15] = (_Float16)f3.w;
#pragma unroll
        for (int nt = 0; nt < 4; ++nt) {
          int col = nt * 16 + r;
          v16h bv;
#pragma unroll
          for (int j = 0; j < 8; ++j) bv[j]     = lVt[col][hl * 8 + j];
#pragma unroll
          for (int j = 0; j < 8; ++j) bv[8 + j] = lVt[col][16 + hl * 8 + j];
          acc[ai][nt] = __builtin_amdgcn_wmma_f32_16x16x32_f16(
              false, av, false, bv, (short)0, acc[ai][nt], false, false);
        }
      }
    }
    __syncthreads();
  }
  // ---- epilogue: per-wave LDS transpose, coalesced float4 stores ----
  // Wave-private slab + in-order per-wave LDS ops => no barrier needed
  // (a barrier here would be divergent across waves anyway).
#pragma unroll
  for (int ai = 0; ai < 2; ++ai) {
    int mt = wave + ai * 8;
    if (mt >= ntiles) continue;
#pragma unroll
    for (int nt = 0; nt < 4; ++nt)
#pragma unroll
      for (int q = 0; q < 8; ++q)
        sOut[wave][hl * 8 + q][nt * 16 + r] = acc[ai][nt][q];
    int c4 = lane & 15, r2 = lane >> 4;        // 16 lanes/row, 2 rows/iter
#pragma unroll
    for (int it = 0; it < 8; ++it) {
      int row = it * 2 + r2;
      int m = mt * 16 + row;
      if (m < Lq) {
        float4 v = *(const float4*)&sOut[wave][row][c4 * 4];
        *(float4*)(Ob + (size_t)m * oM + c4 * 4) = v;
      }
    }
  }
}

// ---------------------------------------------------------------------------
// LayerNorm over 512 channels. One block (128 thr, float4/thr) per token.
// ---------------------------------------------------------------------------
__device__ __forceinline__ void ln_stats(const float4& x, float* rs, float* rq,
                                         int tid, float& mean, float& rstd)
{
  rs[tid] = x.x + x.y + x.z + x.w;
  rq[tid] = x.x * x.x + x.y * x.y + x.z * x.z + x.w * x.w;
  __syncthreads();
  for (int o = 64; o > 0; o >>= 1) {
    if (tid < o) { rs[tid] += rs[tid + o]; rq[tid] += rq[tid + o]; }
    __syncthreads();
  }
  mean = rs[0] * (1.f / 512.f);
  float var = rq[0] * (1.f / 512.f) - mean * mean;
  rstd = rsqrtf(var + 1e-5f);
}

__global__ __launch_bounds__(128) void ln_merged(
    const float* __restrict__ in, const float* __restrict__ gam,
    const float* __restrict__ bet, float* __restrict__ out)
{
  int tok = blockIdx.x, tid = threadIdx.x;
  float4 x = ((const float4*)in)[(long long)tok * 128 + tid];
  __shared__ float rs[128], rq[128];
  float mean, rstd;
  ln_stats(x, rs, rq, tid, mean, rstd);
  float4 g = ((const float4*)gam)[tid], bb = ((const float4*)bet)[tid];
  float4 y;
  y.x = (x.x - mean) * rstd * g.x + bb.x;
  y.y = (x.y - mean) * rstd * g.y + bb.y;
  y.z = (x.z - mean) * rstd * g.z + bb.z;
  y.w = (x.w - mean) * rstd * g.w + bb.w;
  ((float4*)out)[(long long)tok * 128 + tid] = y;
}

// LayerNorm + restore to [B,H,T,S,64], written into d_out row (b) at colOff.
__global__ __launch_bounds__(128) void ln_restored(
    const float* __restrict__ in, const float* __restrict__ gam,
    const float* __restrict__ bet, float* __restrict__ out,
    int T, int S, long long rowStride, long long colOff)
{
  int tok = blockIdx.x, tid = threadIdx.x;
  float4 x = ((const float4*)in)[(long long)tok * 128 + tid];
  __shared__ float rs[128], rq[128];
  float mean, rstd;
  ln_stats(x, rs, rq, tid, mean, rstd);
  float4 g = ((const float4*)gam)[tid], bb = ((const float4*)bet)[tid];
  float4 y;
  y.x = (x.x - mean) * rstd * g.x + bb.x;
  y.y = (x.y - mean) * rstd * g.y + bb.y;
  y.z = (x.z - mean) * rstd * g.z + bb.z;
  y.w = (x.w - mean) * rstd * g.w + bb.w;
  int b = tok / (T * S); int rr = tok % (T * S);
  int t = rr / S; int s = rr % S;
  int ch = tid * 4; int h = ch >> 6; int c0 = ch & 63;
  long long o = (long long)b * rowStride + colOff +
                ((long long)(h * T + t) * S + s) * 64 + c0;
  ((float4*)(out + o))[0] = y;
}

// ===========================================================================
extern "C" void kernel_launch(void* const* d_in, const int* in_sizes, int n_in,
                              void* d_out, int out_size, void* d_ws,
                              size_t ws_size, hipStream_t stream)
{
  (void)in_sizes; (void)n_in; (void)out_size; (void)ws_size;
  const float* x1 = (const float*)d_in[0];
  const float* x2 = (const float*)d_in[1];
  const float* p1 = (const float*)d_in[2];
  const float* p2 = (const float*)d_in[3];
  float* outp = (float*)d_out;

  // ---- workspace layout (256B aligned slabs) ----
  char* base = (char*)d_ws;
  size_t off = 0;
  auto alloc = [&](size_t bytes) -> void* {
    void* p = base + off;
    off += (bytes + 255) & ~(size_t)255;
    return p;
  };
  const size_t TOKMAX = 31360;                       // B*8*245 == B*40*49
  const size_t TB = TOKMAX * 512 * sizeof(float);    // 64.2 MB token slab
  _Float16* w16 = (_Float16*)alloc(18ull * 512 * 512 * sizeof(_Float16));
  float* x1cm = (float*)alloc(TB);
  float* x2cm = (float*)alloc(TB);
  float* p1m  = (float*)alloc(6272ull * 512 * 4);
  float* p2m  = (float*)alloc(6272ull * 512 * 4);
  float* gQ   = (float*)alloc(TB);
  float* gK   = (float*)alloc(TB);
  float* gV   = (float*)alloc(TB);
  float* gA   = (float*)alloc(TB);
  float* hat1 = (float*)alloc(6272ull * 512 * 4);
  float* hat2 = (float*)alloc(6272ull * 512 * 4);
  float* cat12 = (float*)alloc(12544ull * 512 * 4);
  float* cat21 = (float*)alloc(12544ull * 512 * 4);
  float* X12  = (float*)alloc(TB);
  float* X21  = (float*)alloc(TB);
  float* qp   = (float*)alloc(16ull * 245 * 512 * 4);
  float* kp   = (float*)alloc(16ull * 245 * 512 * 4);
  float* Pb   = (float*)alloc(16ull * 8 * 245 * 256 * 4);  // Lk padded to x32

  auto cdiv = [](long long a, long long b) { return (int)((a + b - 1) / b); };

  // ---- convert all 18 weights to f16 ----
  for (int k = 0; k < 18; ++k)
    cvt_f16<<<cdiv(262144, 256), 256, 0, stream>>>(
        (const float*)d_in[4 + 2 * k], w16 + (size_t)k * 262144, 262144);

  // ---- build merged inputs ----
  {
    long long t4 = 16ll * 8 * 245 * 128;  // x1c = [x1 | p1] along S
    pack_merge<<<cdiv(t4, 256), 256, 0, stream>>>(x1, p1, x1cm, 8, 245, 1, 196, t4);
    t4 = 16ll * 40 * 49 * 128;            // x2c = [x2 ; p2] along T
    pack_merge<<<cdiv(t4, 256), 256, 0, stream>>>(x2, p2, x2cm, 40, 49, 0, 32, t4);
    t4 = 16ll * 8 * 49 * 128;
    pack_merge<<<cdiv(t4, 256), 256, 0, stream>>>(p1, p1, p1m, 8, 49, 1, 49, t4);
    pack_merge<<<cdiv(t4, 256), 256, 0, stream>>>(p2, p2, p2m, 8, 49, 1, 49, t4);
  }

  auto gemm = [&](const float* A, int k, float* C, int M) {
    dim3 grid(cdiv(M, 128), 8);
    gemm512<<<grid, 256, 0, stream>>>(A, w16 + (size_t)k * 262144,
                                      (const float*)d_in[5 + 2 * k], C, M);
  };
  auto pool = [&](const float* in, float* out, int T, int S, int overT) {
    long long t4 = 16ll * (overT ? S : T) * 128;
    pool_mean<<<cdiv(t4, 256), 256, 0, stream>>>(in, out, T, S, overT, t4);
  };
  auto scores = [&](int Lq, int Lk) {
    int Lk32 = (Lk + 31) & ~31;
    attn_scores<<<16 * NH * Lq, 256, 0, stream>>>(qp, kp, Pb, Lq, Lk, Lk32);
  };
  auto apply = [&](int G, int Lq, int Lk, long long vOffB, long long vOffG,
                   long long vK, long long oOffB, long long oOffG, long long oM) {
    int Lk32 = (Lk + 31) & ~31;
    attn_apply<<<16 * G * NH, 256, 0, stream>>>(Pb, gV, gA, G, Lq, Lk, Lk32,
                                                vOffB, vOffG, vK, oOffB, oOffG, oM);
  };
  auto ln = [&](float* dst, int ntok, int j) {
    ln_merged<<<ntok, 128, 0, stream>>>(gA, (const float*)d_in[40 + 2 * j],
                                        (const float*)d_in[41 + 2 * j], dst);
  };

  // ======== Stage 1, stream 1 ========
  gemm(p1m,  0, gQ,  6272);          // Q1_1(p1)
  gemm(x1cm, 1, gK, 31360);          // K1_1(x1c)
  gemm(x1cm, 2, gV, 31360);          // V1_1(x1c)
  pool(gQ, qp, 8, 49, 1);            // -> [B,49,512]
  pool(gK, kp, 8, 245, 1);           // -> [B,245,512]
  scores(49, 245);
  apply(8, 49, 245, 8 * 245 * 512, 245 * 512, 512,
        8 * 49 * 512, 49 * 512, 512);
  ln(hat1, 6272, 0);                 // norm1_1 -> x1_hat [B,8,49,512]

  // ======== Stage 1, stream 2 ========
  gemm(p2m,  3, gQ,  6272);          // Q2_1(p2)
  gemm(x2cm, 4, gK, 31360);          // K2_1(x2c)
  gemm(x2cm, 5, gV, 31360);          // V2_1(x2c)
  pool(gQ, qp, 8, 49, 0);            // pool over S -> [B,8,512]
  pool(gK, kp, 40, 49, 0);           // -> [B,40,512]
  scores(8, 40);
  apply(49, 8, 40, 40 * 49 * 512, 512, 49 * 512,
        8 * 49 * 512, 512, 49 * 512);
  ln(hat2, 6272, 1);                 // norm2_1 -> x2_hat [B,8,49,512]

  // ======== concats ========
  {
    long long t4 = 16ll * 8 * 49 * 128;
    copy_tokens<<<cdiv(t4, 256), 256, 0, stream>>>(hat1, cat12, 8, 49, 8, 98, 0, 0, t4);
    copy_tokens<<<cdiv(t4, 256), 256, 0, stream>>>(hat2, cat12, 8, 49, 8, 98, 0, 49, t4);
    copy_tokens<<<cdiv(t4, 256), 256, 0, stream>>>(hat2, cat21, 8, 49, 16, 49, 0, 0, t4);
    copy_tokens<<<cdiv(t4, 256), 256, 0, stream>>>(hat1, cat21, 8, 49, 16, 49, 8, 0, t4);
  }

  // ======== Stage 2, stream 1 ========
  gemm(x1cm,  6, gQ, 31360);         // Q1_2(x1c)
  gemm(cat12, 7, gK, 12544);         // K1_2(x12_hat) [B,8,98,512]
  gemm(cat12, 8, gV, 12544);         // V1_2(x12_hat)
  pool(gQ, qp, 8, 245, 1);           // -> [B,245,512]
  pool(gK, kp, 8, 98, 1);            // -> [B,98,512]
  scores(245, 98);
  apply(8, 245, 98, 8 * 98 * 512, 98 * 512, 512,
        8 * 245 * 512, 245 * 512, 512);
  ln(X12, 31360, 2);                 // norm1_2 -> x12 [B,8,245,512]

  // ======== Stage 2, stream 2 ========
  gemm(x2cm,   9, gQ, 31360);        // Q2_2(x2c)
  gemm(cat21, 10, gK, 12544);        // K2_2(x21_hat) [B,16,49,512]
  gemm(cat21, 11, gV, 12544);        // V2_2(x21_hat)
  pool(gQ, qp, 40, 49, 0);           // -> [B,40,512]
  pool(gK, kp, 16, 49, 0);           // -> [B,16,512]
  scores(40, 16);
  apply(49, 40, 16, 16 * 49 * 512, 512, 49 * 512,
        40 * 49 * 512, 512, 49 * 512);
  ln(X21, 31360, 3);                 // norm2_2 -> x21 [B,40,49,512]

  // ======== Stage 3, stream 1 ========
  gemm(x1cm, 12, gQ, 31360);         // Q1_3(x1c)
  gemm(X12,  13, gK, 31360);         // K1_3(x12)
  gemm(X12,  14, gV, 31360);         // V1_3(x12)
  pool(gQ, qp, 8, 245, 1);
  pool(gK, kp, 8, 245, 1);
  scores(245, 245);
  apply(8, 245, 245, 8 * 245 * 512, 245 * 512, 512,
        8 * 245 * 512, 245 * 512, 512);
  ln_restored<<<31360, 128, 0, stream>>>(gA, (const float*)d_in[48],
                                         (const float*)d_in[49], outp,
                                         8, 245, 2007040ll, 0ll);

  // ======== Stage 3, stream 2 ========
  gemm(x2cm, 15, gQ, 31360);         // Q2_3(x2c)
  gemm(X21,  16, gK, 31360);         // K2_3(x21)
  gemm(X21,  17, gV, 31360);         // V2_3(x21)
  pool(gQ, qp, 40, 49, 0);
  pool(gK, kp, 40, 49, 0);
  scores(40, 40);
  apply(49, 40, 40, 40 * 49 * 512, 512, 49 * 512,
        40 * 49 * 512, 512, 49 * 512);
  ln_restored<<<31360, 128, 0, stream>>>(gA, (const float*)d_in[50],
                                         (const float*)d_in[51], outp,
                                         40, 49, 2007040ll, 1003520ll);
}
